// MolGNN_29712583754389
// MI455X (gfx1250) — matrix operand order repaired and checked by reading, not compiled, not run on hardware
//
#include <hip/hip_runtime.h>
#include <hip/hip_bf16.h>

#define N_NODES   100000
#define N_EDGES   600000
#define HIDDEN    128
#define OUT_DIM   768
#define LAYERS    3
#define N_GRAPHS  4096
#define NODE_VOCAB 119
#define EDGE_VOCAB 22

typedef __attribute__((ext_vector_type(16))) __bf16 v16bf;
typedef __attribute__((ext_vector_type(8)))  float  v8f;

// ---------------- WMMA fragment helpers (CDNA5 wave32 layouts) ----------------
// A (16-bit, 16x32): lanes 0-15 row=lane, K-octet base 0; lanes 16-31 row=lane-16, +8.
// elems 0..7 -> K = k0 + 8*half + e ; elems 8..15 -> K = k0 + 16 + 8*half + (e-8)
__device__ __forceinline__ v16bf load_a_frag_f32(const float* __restrict__ A, int lda,
                                                 int m0, int k0, int lane) {
    int row = m0 + (lane & 15);
    int h8  = (lane >> 4) << 3;
    const float* p = A + (size_t)row * lda + k0 + h8;
    v16bf a;
#pragma unroll
    for (int e = 0; e < 8; ++e) a[e] = (__bf16)p[e];
#pragma unroll
    for (int e = 0; e < 8; ++e) a[8 + e] = (__bf16)p[16 + e];
    return a;
}

__device__ __forceinline__ v16bf load_a_frag_bf16(const __bf16* __restrict__ A, int lda,
                                                  int m0, int k0, int lane) {
    int row = m0 + (lane & 15);
    int h8  = (lane >> 4) << 3;
    const __bf16* p = A + (size_t)row * lda + k0 + h8;
    v16bf a;
#pragma unroll
    for (int e = 0; e < 8; ++e) a[e] = p[e];
#pragma unroll
    for (int e = 0; e < 8; ++e) a[8 + e] = p[16 + e];
    return a;
}

// ---------------- weight fragment pre-pack ----------------
// Packs row-major f32 W[K x N] into per-(kt,nt,lane) contiguous 16-bf16 B fragments.
// B (16-bit, 32x16): lane = col; elems 0..7 -> K = 8*half + e, elems 8..15 -> K = 16 + 8*half + (e-8);
// element pair (2v,2v+1) = adjacent K-pair.
__global__ void MolGNN_pack_b(const float* __restrict__ W, __bf16* __restrict__ dst,
                              int K, int N) {
    int t = blockIdx.x * blockDim.x + threadIdx.x;        // one thread per (frag, lane)
    int nnt = N >> 4;
    int total = (K >> 5) * nnt * 32;
    if (t >= total) return;
    int lane = t & 31;
    int frag = t >> 5;
    int nt = frag % nnt;
    int kt = frag / nnt;
    int col = (nt << 4) + (lane & 15);
    int kh  = (lane >> 4) << 3;
    __bf16* d = dst + (size_t)t * 16;                     // 32B, 32B-aligned
#pragma unroll
    for (int v = 0; v < 8; ++v) {
        int k = (kt << 5) + kh + 2 * v + ((v >> 2) << 3);
        d[2 * v]     = (__bf16)W[(size_t)k * N + col];
        d[2 * v + 1] = (__bf16)W[(size_t)(k + 1) * N + col];
    }
}

// fragment fetch: one 32-byte vector load per lane (coalesced 1KB per wave)
__device__ __forceinline__ v16bf fetch_b(const __bf16* __restrict__ Wp, int nnt,
                                         int kt, int nt, int lane) {
    const v16bf* f = (const v16bf*)Wp;
    return f[((size_t)kt * nnt + nt) * 32 + lane];
}

// ---------------- small utility kernels ----------------
__global__ void MolGNN_zero(float* __restrict__ p, int n) {
    int i = blockIdx.x * blockDim.x + threadIdx.x;
    if (i < n) p[i] = 0.0f;
}

// Fused: h[n] = sum_f node_tables[f, x[n,f]];  z0[n] = (1+eps[0]) * h[n]
// 32 lanes per node, float4 per lane.
__global__ void __launch_bounds__(128) MolGNN_node_embed(
        const int* __restrict__ x, const float* __restrict__ tables,
        const float* __restrict__ eps, float* __restrict__ h, float* __restrict__ z0) {
    int t  = blockIdx.x * 128 + threadIdx.x;
    int n  = t >> 5;
    int c4 = (t & 31) << 2;
    if (n >= N_NODES) return;
    float ax = 0.f, ay = 0.f, az = 0.f, aw = 0.f;
#pragma unroll
    for (int f = 0; f < 9; ++f) {
        int idx = x[n * 9 + f];
        const float4 tv = *(const float4*)&tables[((size_t)(f * NODE_VOCAB + idx)) * HIDDEN + c4];
        ax += tv.x; ay += tv.y; az += tv.z; aw += tv.w;
    }
    size_t o = (size_t)n * HIDDEN + c4;
    *(float4*)&h[o] = make_float4(ax, ay, az, aw);
    float s0 = 1.0f + eps[0];
    *(float4*)&z0[o] = make_float4(s0 * ax, s0 * ay, s0 * az, s0 * aw);
}

// per-edge: z[dst] += relu(h[src] + edge_emb(e)); edge_emb recomputed from L2-resident
// tables. 32 lanes per edge, float4 per lane, 4 component atomics.
__global__ void __launch_bounds__(128) MolGNN_edge_scatter(
        const float* __restrict__ h, const int* __restrict__ edge_attr,
        const int* __restrict__ edge_index, const float* __restrict__ etab,
        float* __restrict__ z) {
    int t  = blockIdx.x * 128 + threadIdx.x;
    int e  = t >> 5;
    int c4 = (t & 31) << 2;
    if (e >= N_EDGES) return;
    int s = edge_index[e];
    int d = edge_index[N_EDGES + e];
    float ex = 0.f, ey = 0.f, ez = 0.f, ew = 0.f;
#pragma unroll
    for (int f = 0; f < 3; ++f) {
        int idx = edge_attr[e * 3 + f];
        const float4 tv = *(const float4*)&etab[((size_t)(f * EDGE_VOCAB + idx)) * HIDDEN + c4];
        ex += tv.x; ey += tv.y; ez += tv.z; ew += tv.w;
    }
    const float4 hv = *(const float4*)&h[(size_t)s * HIDDEN + c4];
    float mx = fmaxf(hv.x + ex, 0.f);
    float my = fmaxf(hv.y + ey, 0.f);
    float mz = fmaxf(hv.z + ez, 0.f);
    float mw = fmaxf(hv.w + ew, 0.f);
    float* zp = &z[(size_t)d * HIDDEN + c4];
    atomicAdd(zp + 0, mx);
    atomicAdd(zp + 1, my);
    atomicAdd(zp + 2, mz);
    atomicAdd(zp + 3, mw);
}

// Fused GINE MLP. One wave owns a 32x128 node tile (two 16-row WMMA tiles so every
// B fragment load feeds two WMMAs). Intermediate lives in LDS bf16, never hits HBM.
// H = relu( relu(Z*W1 + b1) * W2 + b2 );  optionally also emits znext = (1+eps[nl])*H
// (same-row in-place reuse of Z is race-free: each block reads only its own rows,
//  into registers, before writing).
__global__ void __launch_bounds__(32) MolGNN_mlp(
        const float* __restrict__ Z,
        const __bf16* __restrict__ W1p, const float* __restrict__ b1,
        const __bf16* __restrict__ W2p, const float* __restrict__ b2,
        float* __restrict__ H, const float* __restrict__ eps, int next_layer,
        float* __restrict__ znext) {
    __shared__ __bf16 mid[32 * HIDDEN];
    int lane  = threadIdx.x;
    int m0    = blockIdx.x * 32;
    int col   = lane & 15;
    int rbase = (lane >> 4) << 3;   // C/D row base: 0 or 8
    float sn = znext ? (1.0f + eps[next_layer]) : 0.0f;

    v16bf a[2][4];
#pragma unroll
    for (int rt = 0; rt < 2; ++rt)
#pragma unroll
        for (int kt = 0; kt < 4; ++kt)
            a[rt][kt] = load_a_frag_f32(Z, HIDDEN, m0 + rt * 16, kt * 32, lane);

#pragma unroll
    for (int nt = 0; nt < 8; ++nt) {
        v8f acc0 = {}, acc1 = {};
#pragma unroll
        for (int kt = 0; kt < 4; ++kt) {
            v16bf bf = fetch_b(W1p, 8, kt, nt, lane);
            acc0 = __builtin_amdgcn_wmma_f32_16x16x32_bf16(false, a[0][kt], false, bf,
                                                           (short)0, acc0, false, false);
            acc1 = __builtin_amdgcn_wmma_f32_16x16x32_bf16(false, a[1][kt], false, bf,
                                                           (short)0, acc1, false, false);
        }
        int c = nt * 16 + col;
        float bias = b1[c];
#pragma unroll
        for (int v = 0; v < 8; ++v) {
            mid[(rbase + v) * HIDDEN + c]      = (__bf16)fmaxf(acc0[v] + bias, 0.0f);
            mid[(16 + rbase + v) * HIDDEN + c] = (__bf16)fmaxf(acc1[v] + bias, 0.0f);
        }
    }
    __syncthreads();

    v16bf a2[2][4];
#pragma unroll
    for (int rt = 0; rt < 2; ++rt)
#pragma unroll
        for (int kt = 0; kt < 4; ++kt)
            a2[rt][kt] = load_a_frag_bf16(mid, HIDDEN, rt * 16, kt * 32, lane);

#pragma unroll
    for (int nt = 0; nt < 8; ++nt) {
        v8f acc0 = {}, acc1 = {};
#pragma unroll
        for (int kt = 0; kt < 4; ++kt) {
            v16bf bf = fetch_b(W2p, 8, kt, nt, lane);
            acc0 = __builtin_amdgcn_wmma_f32_16x16x32_bf16(false, a2[0][kt], false, bf,
                                                           (short)0, acc0, false, false);
            acc1 = __builtin_amdgcn_wmma_f32_16x16x32_bf16(false, a2[1][kt], false, bf,
                                                           (short)0, acc1, false, false);
        }
        int c = nt * 16 + col;
        float bias = b2[c];
#pragma unroll
        for (int v = 0; v < 8; ++v) {
            size_t o0 = (size_t)(m0 + rbase + v) * HIDDEN + c;
            size_t o1 = (size_t)(m0 + 16 + rbase + v) * HIDDEN + c;
            float h0 = fmaxf(acc0[v] + bias, 0.0f);
            float h1 = fmaxf(acc1[v] + bias, 0.0f);
            H[o0] = h0;
            H[o1] = h1;
            if (znext) { znext[o0] = sn * h0; znext[o1] = sn * h1; }
        }
    }
}

// global_add_pool: g[batch[n]] += h[n]; 32 lanes per node, float4 per lane.
__global__ void __launch_bounds__(128) MolGNN_pool(
        const float* __restrict__ h, const int* __restrict__ batch_ids,
        float* __restrict__ g) {
    int t  = blockIdx.x * 128 + threadIdx.x;
    int n  = t >> 5;
    int c4 = (t & 31) << 2;
    if (n >= N_NODES) return;
    int b = batch_ids[n];
    const float4 hv = *(const float4*)&h[(size_t)n * HIDDEN + c4];
    float* gp = &g[(size_t)b * HIDDEN + c4];
    atomicAdd(gp + 0, hv.x);
    atomicAdd(gp + 1, hv.y);
    atomicAdd(gp + 2, hv.z);
    atomicAdd(gp + 3, hv.w);
}

// out[32x16 tile] = G * Wp + bp  (K=128 via bf16 WMMAs; two row-tiles share B frags)
__global__ void __launch_bounds__(32) MolGNN_proj(
        const float* __restrict__ G, const __bf16* __restrict__ Wpp,
        const float* __restrict__ bp, float* __restrict__ out) {
    int lane = threadIdx.x;
    int m0 = blockIdx.x * 32;
    int nt = blockIdx.y;            // 0..47

    v16bf a[2][4];
#pragma unroll
    for (int rt = 0; rt < 2; ++rt)
#pragma unroll
        for (int kt = 0; kt < 4; ++kt)
            a[rt][kt] = load_a_frag_f32(G, HIDDEN, m0 + rt * 16, kt * 32, lane);

    v8f acc0 = {}, acc1 = {};
#pragma unroll
    for (int kt = 0; kt < 4; ++kt) {
        v16bf bf = fetch_b(Wpp, OUT_DIM >> 4, kt, nt, lane);
        acc0 = __builtin_amdgcn_wmma_f32_16x16x32_bf16(false, a[0][kt], false, bf,
                                                       (short)0, acc0, false, false);
        acc1 = __builtin_amdgcn_wmma_f32_16x16x32_bf16(false, a[1][kt], false, bf,
                                                       (short)0, acc1, false, false);
    }
    int c = nt * 16 + (lane & 15);
    float bias = bp[c];
    int rbase = (lane >> 4) << 3;
#pragma unroll
    for (int v = 0; v < 8; ++v) {
        out[(size_t)(m0 + rbase + v) * OUT_DIM + c]      = acc0[v] + bias;
        out[(size_t)(m0 + 16 + rbase + v) * OUT_DIM + c] = acc1[v] + bias;
    }
}

// in-place row L2-normalize of [N_GRAPHS, OUT_DIM]
__global__ void __launch_bounds__(256) MolGNN_normalize(float* __restrict__ out) {
    __shared__ float red[256];
    int row = blockIdx.x, t = threadIdx.x;
    float ss = 0.0f;
    for (int c = t; c < OUT_DIM; c += 256) {
        float v = out[(size_t)row * OUT_DIM + c];
        ss += v * v;
    }
    red[t] = ss;
    __syncthreads();
    for (int s = 128; s > 0; s >>= 1) {
        if (t < s) red[t] += red[t + s];
        __syncthreads();
    }
    float inv = 1.0f / fmaxf(sqrtf(red[0]), 1e-12f);
    for (int c = t; c < OUT_DIM; c += 256) {
        out[(size_t)row * OUT_DIM + c] *= inv;
    }
}

// ---------------- launch ----------------
extern "C" void kernel_launch(void* const* d_in, const int* in_sizes, int n_in,
                              void* d_out, int out_size, void* d_ws, size_t ws_size,
                              hipStream_t stream) {
    const int*   x           = (const int*)  d_in[0];
    const int*   edge_attr   = (const int*)  d_in[1];
    const int*   edge_index  = (const int*)  d_in[2];
    const int*   batch_ids   = (const int*)  d_in[3];
    const float* node_tables = (const float*)d_in[4];
    const float* edge_tables = (const float*)d_in[5];
    const float* W1          = (const float*)d_in[6];
    const float* b1          = (const float*)d_in[7];
    const float* W2          = (const float*)d_in[8];
    const float* b2          = (const float*)d_in[9];
    const float* eps         = (const float*)d_in[10];
    const float* Wp          = (const float*)d_in[11];
    const float* bp          = (const float*)d_in[12];
    float* out = (float*)d_out;

    char* ws = (char*)d_ws;
    const size_t HB = (size_t)N_NODES * HIDDEN * sizeof(float);        // 51.2 MB
    const size_t WMAT = (size_t)HIDDEN * HIDDEN;                       // 16384 elems
    float*  h    = (float*)(ws);
    float*  z    = (float*)(ws + HB);
    float*  g    = (float*)(ws + 2 * HB);
    char*   wptr = ws + 2 * HB + (size_t)N_GRAPHS * HIDDEN * sizeof(float);
    __bf16* w1p  = (__bf16*)wptr;                       // LAYERS * 32KB, fragment-packed
    __bf16* w2p  = w1p + LAYERS * WMAT;
    __bf16* wpp  = w2p + LAYERS * WMAT;                 // 192KB, fragment-packed

    // pack weights into WMMA B-fragment order (done per launch; deterministic)
    const int nfragW  = (HIDDEN / 32) * (HIDDEN / 16) * 32;   // 1024 threads / matrix
    const int nfragWp = (HIDDEN / 32) * (OUT_DIM / 16) * 32;  // 6144 threads
    for (int l = 0; l < LAYERS; ++l) {
        MolGNN_pack_b<<<(nfragW + 255) / 256, 256, 0, stream>>>(
                W1 + l * WMAT, w1p + l * WMAT, HIDDEN, HIDDEN);
        MolGNN_pack_b<<<(nfragW + 255) / 256, 256, 0, stream>>>(
                W2 + l * WMAT, w2p + l * WMAT, HIDDEN, HIDDEN);
    }
    MolGNN_pack_b<<<(nfragWp + 255) / 256, 256, 0, stream>>>(Wp, wpp, HIDDEN, OUT_DIM);

    // embed (+ fused z0 init): 32 lanes/node
    MolGNN_node_embed<<<(N_NODES * 32) / 128, 128, 0, stream>>>(x, node_tables, eps, h, z);

    for (int l = 0; l < LAYERS; ++l) {
        // z currently holds (1+eps[l])*h; accumulate messages into it
        MolGNN_edge_scatter<<<(N_EDGES * 32) / 128, 128, 0, stream>>>(
                h, edge_attr, edge_index, edge_tables, z);
        // MLP writes h, and (fused) re-seeds z = (1+eps[l+1])*h for the next layer
        float* znext = (l + 1 < LAYERS) ? z : nullptr;
        MolGNN_mlp<<<N_NODES / 32, 32, 0, stream>>>(z,
                w1p + l * WMAT, b1 + l * HIDDEN,
                w2p + l * WMAT, b2 + l * HIDDEN, h, eps, l + 1, znext);
    }

    MolGNN_zero<<<(N_GRAPHS * HIDDEN) / 256, 256, 0, stream>>>(g, N_GRAPHS * HIDDEN);
    MolGNN_pool<<<(N_NODES * 32) / 128, 128, 0, stream>>>(h, batch_ids, g);
    MolGNN_proj<<<dim3(N_GRAPHS / 32, OUT_DIM / 16), 32, 0, stream>>>(g, wpp, bp, out);
    MolGNN_normalize<<<N_GRAPHS, 256, 0, stream>>>(out);
}
